// Head_36026185679213
// MI455X (gfx1250) — matrix operand order repaired and checked by reading, not compiled, not run on hardware
//
#include <hip/hip_runtime.h>
#include <hip/hip_bf16.h>

typedef __attribute__((ext_vector_type(16))) _Float16 v16h;
typedef __attribute__((ext_vector_type(8)))  _Float16 h8;
typedef __attribute__((ext_vector_type(8)))  float    v8f;

#define B_SZ 4
#define T_SZ 4096
#define C_SZ 1024
#define HD   64
#define SCALE 0.03125f               // 1024^-0.5
#define SLOPE 0.70710678118654752f   // 2^-0.5 (head 0 of 16, alibi base 2^8)

__device__ __forceinline__ v8f wmma_f16(v16h a, v16h b, v8f c) {
  // D = A(16x32 f16) * B(32x16 f16) + C(16x16 f32)
  return __builtin_amdgcn_wmma_f32_16x16x32_f16(false, a, false, b, (short)0, c,
                                                false, false);
}

__device__ __forceinline__ v16h cat16(h8 lo, h8 hi) {
  return __builtin_shufflevector(lo, hi, 0, 1, 2, 3, 4, 5, 6, 7,
                                         8, 9, 10, 11, 12, 13, 14, 15);
}

__device__ __forceinline__ v16h ldfrag(const _Float16* p) {
  return cat16(*(const h8*)p, *(const h8*)(p + 16));
}

// ---------------------------------------------------------------------------
// Kernel 0: one-shot fp32 -> fp16 conversion of the three weight matrices.
// ---------------------------------------------------------------------------
__global__ __launch_bounds__(256) void cvtw_kernel(
    const float* __restrict__ wq, const float* __restrict__ wk,
    const float* __restrict__ wv, _Float16* __restrict__ wh)
{
  const int i = blockIdx.x * 256 + threadIdx.x;      // 65536 elems per matrix
  wh[i]                 = (_Float16)wq[i];
  wh[HD * C_SZ + i]     = (_Float16)wk[i];
  wh[2 * HD * C_SZ + i] = (_Float16)wv[i];
}

// ---------------------------------------------------------------------------
// Kernel 1: QKV projection.  One wave computes a 16-token tile of Q,K,V.
//   Q,K stored [B*T, 64] f16 ;  V stored transposed [B, 64, T] f16.
// B-fragments software-pipelined (4 buffers, prefetch distance 2);
// x raw data prefetched one k-step ahead.
// ---------------------------------------------------------------------------
__global__ __launch_bounds__(256) void qkv_kernel(
    const float* __restrict__ x, const _Float16* __restrict__ wh,
    _Float16* __restrict__ Q, _Float16* __restrict__ Km,
    _Float16* __restrict__ Vt)
{
  const int lane = threadIdx.x & 31;
  const int wave = threadIdx.x >> 5;
  const int tile = blockIdx.x * 8 + wave;          // 1024 tiles of 16 tokens
  const int g0   = tile * 16;
  const int n    = lane & 15;
  const int hi   = lane >> 4;

  v8f c[3][4] = {};                                // [mat][ntile] accumulators
  const float*    xrow  = x + (size_t)(g0 + n) * C_SZ + (hi << 3);
  const _Float16* wbase = wh + (size_t)n * C_SZ + (hi << 3);
  // fragment f (0..11): mat = f>>2, nt = f&3
  //   offset = mat*HD*C_SZ + nt*16*C_SZ + ks*32

  float4 xr[4];                                    // raw x for current k-step
  xr[0] = *(const float4*)(xrow);
  xr[1] = *(const float4*)(xrow + 4);
  xr[2] = *(const float4*)(xrow + 16);
  xr[3] = *(const float4*)(xrow + 20);

  #pragma unroll 1
  for (int ks = 0; ks < C_SZ / 32; ++ks) {
    // ---- convert prefetched x to the A fragment ----
    v16h a;
    a[0]  = (_Float16)xr[0].x; a[1]  = (_Float16)xr[0].y;
    a[2]  = (_Float16)xr[0].z; a[3]  = (_Float16)xr[0].w;
    a[4]  = (_Float16)xr[1].x; a[5]  = (_Float16)xr[1].y;
    a[6]  = (_Float16)xr[1].z; a[7]  = (_Float16)xr[1].w;
    a[8]  = (_Float16)xr[2].x; a[9]  = (_Float16)xr[2].y;
    a[10] = (_Float16)xr[2].z; a[11] = (_Float16)xr[2].w;
    a[12] = (_Float16)xr[3].x; a[13] = (_Float16)xr[3].y;
    a[14] = (_Float16)xr[3].z; a[15] = (_Float16)xr[3].w;
    // ---- prefetch next k-step's x (overlaps entire fragment loop) ----
    if (ks + 1 < C_SZ / 32) {
      const float* xn = xrow + (ks + 1) * 32;
      xr[0] = *(const float4*)(xn);
      xr[1] = *(const float4*)(xn + 4);
      xr[2] = *(const float4*)(xn + 16);
      xr[3] = *(const float4*)(xn + 20);
    }
    // ---- 12 fragment GEMMs, pipelined with prefetch distance 2 ----
    const _Float16* wk0 = wbase + ks * 32;
    v16h bbuf[4];
    bbuf[0] = ldfrag(wk0);                                     // f=0: mat0,nt0
    bbuf[1] = ldfrag(wk0 + 16 * C_SZ);                         // f=1: mat0,nt1
    #pragma unroll
    for (int f = 0; f < 12; ++f) {
      if (f + 2 < 12) {
        const int fn  = f + 2;
        const int mat = fn >> 2, nt = fn & 3;
        bbuf[fn & 3] =
            ldfrag(wk0 + ((size_t)mat * HD + nt * 16) * C_SZ);
      }
      c[f >> 2][f & 3] = wmma_f16(a, bbuf[f & 3], c[f >> 2][f & 3]);
    }
  }

  #pragma unroll
  for (int nt = 0; nt < 4; ++nt) {
    #pragma unroll
    for (int v = 0; v < 8; ++v) {
      const int m  = v + (hi << 3);
      const int g  = g0 + m;
      const int hd = nt * 16 + n;
      Q [(size_t)g * HD + hd] = (_Float16)c[0][nt][v];
      Km[(size_t)g * HD + hd] = (_Float16)c[1][nt][v];
      const int b = g >> 12, t = g & (T_SZ - 1);
      Vt[((size_t)b * HD + hd) * T_SZ + t] = (_Float16)c[2][nt][v];
    }
  }
}

// ---------------------------------------------------------------------------
// Kernel 2: flash attention.  One wave owns a 16-query tile; key blocks of 32.
// Tiles issued largest-q0 first; V fragments loaded before the softmax so
// their latency hides under the VALU work.
// ---------------------------------------------------------------------------
#define ATTN_WAVES 4
__global__ __launch_bounds__(128) void attn_kernel(
    const _Float16* __restrict__ Q, const _Float16* __restrict__ Km,
    const _Float16* __restrict__ Vt, float* __restrict__ out)
{
  __shared__ _Float16 lds[ATTN_WAVES][16 * 32];
  const int lane  = threadIdx.x & 31;
  const int wave  = threadIdx.x >> 5;
  const int tid   = blockIdx.x * ATTN_WAVES + wave;
  const int tile  = (B_SZ * (T_SZ / 16) - 1) - tid;   // longest tiles first
  const int b     = tile >> 8;                        // 256 tiles per batch
  const int q0    = (tile & 255) * 16;
  const int n     = lane & 15;
  const int hi    = lane >> 4;
  _Float16* pbuf  = lds[wave];

  // Q fragments (A layout): per kk, two contiguous 8-half runs
  v16h qa[2];
  {
    const _Float16* qrow = Q + (size_t)(b * T_SZ + q0 + n) * HD;
    #pragma unroll
    for (int kk = 0; kk < 2; ++kk) {
      const _Float16* p = qrow + kk * 32 + (hi << 3);
      qa[kk] = cat16(*(const h8*)p, *(const h8*)(p + 16));
    }
  }

  float m_s[8], l_s[8];
  v8f acc[4] = {};
  #pragma unroll
  for (int v = 0; v < 8; ++v) { m_s[v] = -1e30f; l_s[v] = 0.0f; }

  const int jend = q0 + 15;
  for (int j = 0; j <= jend; j += 32) {
    // ---- S = Q * K^T (two 16x16 tiles covering 32 keys) ----
    v8f s0 = {}, s1 = {};
    #pragma unroll
    for (int kk = 0; kk < 2; ++kk) {
      const _Float16* kr0 =
          Km + (size_t)(b * T_SZ + j + n) * HD + kk * 32 + (hi << 3);
      const _Float16* kr1 = kr0 + 16 * HD;
      v16h kb0 = ldfrag(kr0);
      v16h kb1 = ldfrag(kr1);
      s0 = wmma_f16(qa[kk], kb0, s0);
      s1 = wmma_f16(qa[kk], kb1, s1);
    }

    // ---- V fragments: independent of softmax, load now to hide latency ----
    v16h vb[4];
    #pragma unroll
    for (int nt = 0; nt < 4; ++nt) {
      vb[nt] = ldfrag(Vt + ((size_t)b * HD + nt * 16 + n) * T_SZ + j +
                      (hi << 3));
    }

    // ---- scale + ALiBi + causal mask + online softmax ----
    #pragma unroll
    for (int v = 0; v < 8; ++v) {
      const int i  = q0 + v + (hi << 3);
      const int c0 = j + n, c1 = c0 + 16;
      float e0 = (c0 <= i) ? s0[v] * SCALE + (float)(c0 - i) * SLOPE : -1e30f;
      float e1 = (c1 <= i) ? s1[v] * SCALE + (float)(c1 - i) * SLOPE : -1e30f;
      float t = fmaxf(e0, e1);                // row max within 16-lane half
      t = fmaxf(t, __shfl_xor(t, 1, 32));
      t = fmaxf(t, __shfl_xor(t, 2, 32));
      t = fmaxf(t, __shfl_xor(t, 4, 32));
      t = fmaxf(t, __shfl_xor(t, 8, 32));
      const float mn   = fmaxf(m_s[v], t);
      const float corr = __expf(m_s[v] - mn);
      const float p0 = __expf(e0 - mn);
      const float p1 = __expf(e1 - mn);
      float rs = p0 + p1;                     // row sum within half
      rs += __shfl_xor(rs, 1, 32);
      rs += __shfl_xor(rs, 2, 32);
      rs += __shfl_xor(rs, 4, 32);
      rs += __shfl_xor(rs, 8, 32);
      l_s[v] = l_s[v] * corr + rs;
      m_s[v] = mn;
      #pragma unroll
      for (int nt = 0; nt < 4; ++nt) acc[nt][v] *= corr;
      const int m = v + (hi << 3);
      pbuf[m * 32 + n]      = (_Float16)p0;   // C-layout -> row-major in LDS
      pbuf[m * 32 + n + 16] = (_Float16)p1;
    }
    __builtin_amdgcn_wave_barrier();
    asm volatile("s_wait_dscnt 0" ::: "memory");  // LDS stores visible to loads

    // ---- re-read P in A-fragment layout (two contiguous 8-half runs) ----
    v16h pa;
    {
      const _Float16* p = pbuf + n * 32 + (hi << 3);
      pa = cat16(*(const h8*)p, *(const h8*)(p + 16));
    }
    __builtin_amdgcn_wave_barrier();

    // ---- O += P * V ----
    #pragma unroll
    for (int nt = 0; nt < 4; ++nt) {
      acc[nt] = wmma_f16(pa, vb[nt], acc[nt]);
    }
  }

  // ---- normalize and write fp32 output ----
  #pragma unroll
  for (int nt = 0; nt < 4; ++nt) {
    #pragma unroll
    for (int v = 0; v < 8; ++v) {
      const int m = v + (hi << 3);
      out[(size_t)(b * T_SZ + q0 + m) * HD + nt * 16 + n] = acc[nt][v] / l_s[v];
    }
  }
}

extern "C" void kernel_launch(void* const* d_in, const int* in_sizes, int n_in,
                              void* d_out, int out_size, void* d_ws, size_t ws_size,
                              hipStream_t stream) {
  const float* x  = (const float*)d_in[0];
  const float* wq = (const float*)d_in[1];
  const float* wk = (const float*)d_in[2];
  const float* wv = (const float*)d_in[3];
  float* out = (float*)d_out;

  _Float16* Q  = (_Float16*)d_ws;                       // [B*T, 64] f16
  _Float16* Km = Q  + (size_t)B_SZ * T_SZ * HD;         // [B*T, 64] f16
  _Float16* Vt = Km + (size_t)B_SZ * T_SZ * HD;         // [B, 64, T] f16
  _Float16* WH = Vt + (size_t)B_SZ * HD * T_SZ;         // [3, 64, 1024] f16

  cvtw_kernel<<<(HD * C_SZ) / 256, 256, 0, stream>>>(wq, wk, wv, WH);
  qkv_kernel<<<128, 256, 0, stream>>>(x, WH, Q, Km, Vt);
  attn_kernel<<<(B_SZ * (T_SZ / 16)) / ATTN_WAVES, ATTN_WAVES * 32, 0, stream>>>(
      Q, Km, Vt, out);
}
